// KC_Attention_LSTM_76647986365122
// MI455X (gfx1250) — matrix operand order repaired and checked by reading, not compiled, not run on hardware
//
#include <hip/hip_runtime.h>
#include <cstdint>
#include <cstddef>

// ---------------- CDNA5 WMMA types & fragment helpers (wave32) ----------------
typedef __attribute__((ext_vector_type(16))) __bf16 bf16x16;
typedef __attribute__((ext_vector_type(8)))  __bf16 bf16x8;
typedef __attribute__((ext_vector_type(8)))  float  v8f;

#define WMMA_BF16(a, b, c) \
  __builtin_amdgcn_wmma_f32_16x16x32_bf16(false, (a), false, (b), (short)0, (c), false, false)

// A fragment (16x32 bf16): lane<16 holds M=lane, K = {0..7, 16..23}; lane>=16 holds K = {8..15, 24..31}
__device__ __forceinline__ bf16x16 load_a_frag(const __bf16* rowptr, int kbase, int lane) {
  int ko = (lane >= 16) ? 8 : 0;
  bf16x8 lo = *(const bf16x8*)(rowptr + kbase + ko);
  bf16x8 hi = *(const bf16x8*)(rowptr + kbase + 16 + ko);
  return __builtin_shufflevector(lo, hi, 0,1,2,3,4,5,6,7,8,9,10,11,12,13,14,15);
}
// B fragment (32x16 bf16): lane holds N = lane&15, K = {0..7, 16..23} (lane<16) or {8..15, 24..31} (lane>=16)
__device__ __forceinline__ bf16x16 load_b_frag(const __bf16* colptr, int kbase, int lane) {
  int ko = (lane >= 16) ? 8 : 0;
  bf16x8 lo = *(const bf16x8*)(colptr + kbase + ko);
  bf16x8 hi = *(const bf16x8*)(colptr + kbase + 16 + ko);
  return __builtin_shufflevector(lo, hi, 0,1,2,3,4,5,6,7,8,9,10,11,12,13,14,15);
}

__device__ __forceinline__ float sigf(float x) { return 1.0f / (1.0f + expf(-x)); }

// ---------------- problem constants ----------------
constexpr int cT    = 100;
constexpr int cMCL  = 100;
constexpr int cBT   = 1600;   // B*T
constexpr int cXROW = 1330;
constexpr int cFK   = 512;    // padded K of "full" (500 -> 512)
constexpr int cRNNP = 960;    // padded LSTM input (950 -> 960)
constexpr int cG4   = 512;    // 4*HID
constexpr int cHID  = 128;
constexpr int cFCK  = 384;    // padded FC input (378 -> 384)

// ---------------- weight conversion: f32 [rows][cols] -> bf16 [rpad][cpad] ----------------
__global__ void k_cvt(const float* __restrict__ src, __bf16* __restrict__ dst,
                      int rows, int cols, int cpad, int total) {
  for (int e = blockIdx.x * blockDim.x + threadIdx.x; e < total; e += gridDim.x * blockDim.x) {
    int r = e / cpad, c = e % cpad;
    float v = (r < rows && c < cols) ? src[(size_t)r * cols + c] : 0.0f;
    dst[e] = (__bf16)v;
  }
}
__global__ void k_padf(const float* __restrict__ src, float* __restrict__ dst, int n, int npad) {
  int e = blockIdx.x * blockDim.x + threadIdx.x;
  if (e < npad) dst[e] = (e < n) ? src[e] : 0.0f;
}

// ---------------- K1: per-(b,t) fused gather + 2 WMMA GEMMs + KC attention ----------------
// dyn LDS: fullh (64x512 bf16) | transl (100x112 f32) | cwl[112] | ial[112] | kcl[64] | bvl[112] | awst[8*112] | misc[8]
__global__ __launch_bounds__(256) void
k1(const float* __restrict__ x, const float* __restrict__ node_emb, const float* __restrict__ path_emb,
   const __bf16* __restrict__ WnaB, const float* __restrict__ bna_p, const float* __restrict__ wat_p,
   const float* __restrict__ battn,
   const __bf16* __restrict__ WiaB, const float* __restrict__ bia_p,
   const __bf16* __restrict__ WbB,  const float* __restrict__ bb,
   const float* __restrict__ kc_emb, const float* __restrict__ Wt, const float* __restrict__ bt_b,
   float* __restrict__ cw_g, __bf16* __restrict__ rnn_bf, float* __restrict__ w_g) {
  extern __shared__ char smem_raw[];
  __bf16* fullh = (__bf16*)smem_raw;                    // 64*512 bf16
  float*  transl = (float*)(smem_raw + 64 * 512 * 2);   // 100*112
  float*  cwl  = transl + 100 * 112;
  float*  ial  = cwl + 112;
  float*  kcl  = ial + 112;
  float*  bvl  = kcl + 64;
  float*  awst = bvl + 112;                             // 8*112
  float*  misc = awst + 8 * 112;

  const int ibt = blockIdx.x;
  const float* xrow = x + (size_t)ibt * cXROW;
  const int tid = threadIdx.x, lane = tid & 31, wv = tid >> 5;

  for (int i = tid; i < 112; i += 256) { cwl[i] = 0.0f; ial[i] = 0.0f; }
  for (int i = tid; i < 64; i += 256)  kcl[i] = (i < 50) ? xrow[1280 + i] : 0.0f;
  __syncthreads();

  for (int half = 0; half < 2; ++half) {
    const int r0 = half * 64;
    // gather 64 rows of "full" into LDS as bf16 (cols: node|node|path|rnn_first|pad)
    for (int e = tid; e < 64 * cFK; e += 256) {
      int mr = e >> 9, k = e & 511;
      int m = r0 + mr;
      float v = 0.0f;
      if (m < cMCL && k < 500) {
        if (k < 100)      { int i0 = (int)xrow[200 + 3 * m]; v = node_emb[(size_t)i0 * 100 + k]; }
        else if (k < 200) { int i2 = (int)xrow[202 + 3 * m]; v = node_emb[(size_t)i2 * 100 + (k - 100)]; }
        else if (k < 300) { int i1 = (int)xrow[201 + 3 * m]; v = path_emb[(size_t)i1 * 100 + (k - 200)]; }
        else              { v = xrow[k - 300]; }
      }
      fullh[e] = (__bf16)v;
    }
    __syncthreads();

    const int mtiles = (half == 0) ? 4 : 3;
    // GEMM1: full @ Wna.T (N=512), then tanh(.+bna)*Wattn -> cw partials (ds_add_f32)
    for (int task = wv; task < mtiles * 2; task += 8) {
      int mt = task >> 1, nh = task & 1;
      const __bf16* arow = fullh + (size_t)(mt * 16 + (lane & 15)) * cFK;
      for (int nt = 0; nt < 16; ++nt) {
        int n0 = (nh * 16 + nt) * 16;
        const __bf16* brow = WnaB + (size_t)(n0 + (lane & 15)) * cFK;
        v8f acc = {};
        for (int kt = 0; kt < 16; ++kt) {
          __builtin_prefetch(brow + (kt + 1) * 32, 0, 0);
          bf16x16 a = load_a_frag(arow, kt * 32, lane);
          bf16x16 b = load_b_frag(brow, kt * 32, lane);
          acc = WMMA_BF16(a, b, acc);
        }
        int n = n0 + (lane & 15);
        int mbase = r0 + mt * 16 + ((lane >= 16) ? 8 : 0);
        float wA = wat_p[n], bN = bna_p[n];
        #pragma unroll
        for (int vv = 0; vv < 8; ++vv) {
          float g = tanhf(acc[vv] + bN);
          atomicAdd(&cwl[mbase + vv], g * wA);
        }
      }
    }
    // GEMM2: trans = full @ Wia.T (N=112) -> LDS
    for (int task = wv; task < mtiles * 7; task += 8) {
      int mt = task / 7, nt = task % 7;
      const __bf16* arow = fullh + (size_t)(mt * 16 + (lane & 15)) * cFK;
      const __bf16* brow = WiaB + (size_t)(nt * 16 + (lane & 15)) * cFK;
      v8f acc = {};
      for (int kt = 0; kt < 16; ++kt) {
        bf16x16 a = load_a_frag(arow, kt * 32, lane);
        bf16x16 b = load_b_frag(brow, kt * 32, lane);
        acc = WMMA_BF16(a, b, acc);
      }
      int n = nt * 16 + (lane & 15);
      int mbase = r0 + mt * 16 + ((lane >= 16) ? 8 : 0);
      float bN = bia_p[n];
      #pragma unroll
      for (int vv = 0; vv < 8; ++vv) {
        int m = mbase + vv;
        if (m < cMCL) transl[(size_t)m * 112 + n] = acc[vv] + bN;
      }
    }
    __syncthreads();
  }

  // kcn
  if (tid == 0) {
    float s = 0.0f;
    for (int k = 0; k < 50; ++k) s += kcl[k];
    misc[0] = 1.0f / ((s == 0.0f) ? 1.0f : s);
  }
  __syncthreads();
  const float inv_kcn = misc[0];

  // KC attention: per wave k = wv + 8*kk; scores[k][m] = sum_d kc_emb[k][d]*trans[m][d]
  float iacc[4] = {0, 0, 0, 0};
  for (int kk = 0; kk < 7; ++kk) {
    int k = wv + kk * 8;
    bool act = (k < 50) && (kcl[k] != 0.0f);   // wave-uniform
    if (act) {
      float Rm[4]; float mymax = -1e30f;
      #pragma unroll
      for (int s = 0; s < 4; ++s) {
        int m = lane + s * 32;
        float r = -1e30f;
        if (m < cMCL) {
          r = 0.0f;
          const float* ke = kc_emb + (size_t)k * 100;
          const float* tr = transl + (size_t)m * 112;
          for (int d = 0; d < 100; ++d) r += ke[d] * tr[d];
        }
        Rm[s] = r; mymax = fmaxf(mymax, r);
      }
      for (int off = 16; off; off >>= 1) mymax = fmaxf(mymax, __shfl_xor(mymax, off, 32));
      float es = 0.0f, Em[4];
      #pragma unroll
      for (int s = 0; s < 4; ++s) {
        int m = lane + s * 32;
        Em[s] = (m < cMCL) ? expf(Rm[s] - mymax) : 0.0f;
        es += Em[s];
      }
      for (int off = 16; off; off >>= 1) es += __shfl_xor(es, off, 32);
      float inv = 1.0f / es;
      #pragma unroll
      for (int s = 0; s < 4; ++s) { int m = lane + s * 32; if (m < cMCL) awst[wv * 112 + m] = Em[s] * inv; }
    }
    __builtin_amdgcn_wave_barrier();
    asm volatile("" ::: "memory");
    if (act) {
      #pragma unroll
      for (int s = 0; s < 4; ++s) {
        int d = lane + s * 32;
        if (d < 100) {
          float a = 0.0f;
          for (int m = 0; m < cMCL; ++m) a += awst[wv * 112 + m] * transl[(size_t)m * 112 + d];
          iacc[s] += a * kcl[k];
        }
      }
    }
    __builtin_amdgcn_wave_barrier();
    asm volatile("" ::: "memory");
  }
  #pragma unroll
  for (int s = 0; s < 4; ++s) { int d = lane + s * 32; if (d < 100) atomicAdd(&ial[d], iacc[s]); }
  __syncthreads();

  // finalize ia, bert_vec, rnn_input pieces, cw
  __bf16* rrow = rnn_bf + (size_t)ibt * cRNNP;
  for (int d = tid; d < 100; d += 256) {
    float v = ial[d] * inv_kcn;
    ial[d] = v;
    rrow[250 + d] = (__bf16)v;
  }
  for (int n = tid; n < 112; n += 256) {
    float acc = 0.0f;
    if (n < 100) {
      const __bf16* wb = WbB + (size_t)n * 768;
      for (int j = 0; j < 768; ++j) acc += xrow[500 + j] * (float)wb[j];
      acc += bb[n];
      rrow[850 + n] = (__bf16)acc;
    }
    bvl[n] = acc;
  }
  for (int i = tid; i < 50; i += 256)  rrow[i] = (__bf16)kcl[i];
  for (int i = tid; i < 200; i += 256) rrow[50 + i] = (__bf16)xrow[i];
  for (int i = tid; i < 10; i += 256)  rrow[950 + i] = (__bf16)0.0f;
  for (int m = tid; m < cMCL; m += 256) cw_g[(size_t)ibt * 100 + m] = cwl[m] + battn[0];
  __syncthreads();

  // w = [bert_vec, kc, ia] . Wt + bt
  if (wv == 0) {
    float acc = 0.0f;
    for (int n = lane; n < 100; n += 32) acc += bvl[n] * Wt[n];
    for (int k = lane; k < 50; k += 32)  acc += kcl[k] * Wt[100 + k];
    for (int d = lane; d < 100; d += 32) acc += ial[d] * Wt[150 + d];
    for (int off = 16; off; off >>= 1) acc += __shfl_xor(acc, off, 32);
    if (lane == 0) w_g[ibt] = acc + bt_b[0];
  }
}

// ---------------- K2: aw = softmax over T axis of cw ----------------
__global__ void k_aw(const float* __restrict__ cw, float* __restrict__ aw) {
  int id = blockIdx.x * blockDim.x + threadIdx.x;
  if (id >= cBT) return;
  int b = id / 100, m = id % 100;
  const float* base = cw + (size_t)b * 100 * 100 + m;
  float mx = -1e30f;
  for (int t = 0; t < cT; ++t) mx = fmaxf(mx, base[(size_t)t * 100]);
  float s = 0.0f;
  for (int t = 0; t < cT; ++t) s += expf(base[(size_t)t * 100] - mx);
  float inv = 1.0f / s;
  float* ob = aw + (size_t)b * 100 * 100 + m;
  for (int t = 0; t < cT; ++t) ob[(size_t)t * 100] = expf(base[(size_t)t * 100] - mx) * inv;
}

// ---------------- K3: na = sum_m aw[m]*full[m,:] via weighted embedding regather ----------------
__global__ __launch_bounds__(256) void
k_na(const float* __restrict__ x, const float* __restrict__ node_emb, const float* __restrict__ path_emb,
     const float* __restrict__ aw_g, __bf16* __restrict__ rnn_bf) {
  __shared__ float awv[100];
  __shared__ int idx0[100], idx1[100], idx2[100];
  __shared__ float Ssh;
  int ibt = blockIdx.x, tid = threadIdx.x;
  const float* xrow = x + (size_t)ibt * cXROW;
  for (int m = tid; m < 100; m += 256) {
    awv[m]  = aw_g[(size_t)ibt * 100 + m];
    idx0[m] = (int)xrow[200 + 3 * m];
    idx1[m] = (int)xrow[201 + 3 * m];
    idx2[m] = (int)xrow[202 + 3 * m];
  }
  __syncthreads();
  if (tid == 0) { float s = 0.0f; for (int m = 0; m < 100; ++m) s += awv[m]; Ssh = s; }
  __syncthreads();
  __bf16* rrow = rnn_bf + (size_t)ibt * cRNNP + 350;
  for (int e = tid; e < 500; e += 256) {
    float v = 0.0f;
    if (e < 100)      { for (int m = 0; m < 100; ++m) v += awv[m] * node_emb[(size_t)idx0[m] * 100 + e]; }
    else if (e < 200) { for (int m = 0; m < 100; ++m) v += awv[m] * node_emb[(size_t)idx2[m] * 100 + (e - 100)]; }
    else if (e < 300) { for (int m = 0; m < 100; ++m) v += awv[m] * path_emb[(size_t)idx1[m] * 100 + (e - 200)]; }
    else              { v = xrow[e - 300] * Ssh; }
    rrow[e] = (__bf16)v;
  }
}

// ---------------- K4: LSTM pre-GEMM (1600x960)x(960x512) ----------------
__global__ __launch_bounds__(256) void
k_pre(const __bf16* __restrict__ rnnbf, const __bf16* __restrict__ WihB,
      const float* __restrict__ bih, const float* __restrict__ bhh, float* __restrict__ pre_g) {
  int mt = blockIdx.x, lane = threadIdx.x & 31, wv = threadIdx.x >> 5;
  const __bf16* arow = rnnbf + (size_t)(mt * 16 + (lane & 15)) * cRNNP;
  for (int nt = wv * 4; nt < wv * 4 + 4; ++nt) {
    const __bf16* brow = WihB + (size_t)(nt * 16 + (lane & 15)) * cRNNP;
    v8f acc = {};
    for (int kt = 0; kt < 30; ++kt) {
      __builtin_prefetch(brow + (kt + 1) * 32, 0, 0);
      bf16x16 a = load_a_frag(arow, kt * 32, lane);
      bf16x16 b = load_b_frag(brow, kt * 32, lane);
      acc = WMMA_BF16(a, b, acc);
    }
    int n = nt * 16 + (lane & 15);
    int row0 = mt * 16 + ((lane >= 16) ? 8 : 0);
    float bsum = bih[n] + bhh[n];
    #pragma unroll
    for (int vv = 0; vv < 8; ++vv) pre_g[(size_t)(row0 + vv) * cG4 + n] = acc[vv] + bsum;
  }
}

// ---------------- K5: sequential LSTM scan; per step 16x128x512 WMMA ----------------
__global__ __launch_bounds__(256) void
k_scan(const float* __restrict__ pre_g, const __bf16* __restrict__ WhhB, float* __restrict__ lstm_g) {
  __shared__ __bf16 hbf[16 * 128];
  __shared__ float  cst[16 * 128];
  __shared__ float  gbuf[16 * 512];
  int tid = threadIdx.x, lane = tid & 31, wv = tid >> 5;
  for (int e = tid; e < 2048; e += 256) { hbf[e] = (__bf16)0.0f; cst[e] = 0.0f; }
  __syncthreads();
  for (int t = 0; t < cT; ++t) {
    const __bf16* arow = hbf + (size_t)(lane & 15) * cHID;
    for (int nt = wv * 4; nt < wv * 4 + 4; ++nt) {
      const __bf16* brow = WhhB + (size_t)(nt * 16 + (lane & 15)) * cHID;
      v8f acc = {};
      for (int kt = 0; kt < 4; ++kt) {
        bf16x16 a = load_a_frag(arow, kt * 32, lane);
        bf16x16 b = load_b_frag(brow, kt * 32, lane);
        acc = WMMA_BF16(a, b, acc);
      }
      int n = nt * 16 + (lane & 15);
      int b0 = (lane >= 16) ? 8 : 0;
      #pragma unroll
      for (int vv = 0; vv < 8; ++vv)
        gbuf[(size_t)(b0 + vv) * cG4 + n] = acc[vv] + pre_g[(size_t)((b0 + vv) * cT + t) * cG4 + n];
    }
    __syncthreads();
    for (int e = tid; e < 2048; e += 256) {
      int bb = e >> 7, j = e & 127;
      float gi = gbuf[(size_t)bb * cG4 + j];
      float gf = gbuf[(size_t)bb * cG4 + 128 + j];
      float gg = gbuf[(size_t)bb * cG4 + 256 + j];
      float go = gbuf[(size_t)bb * cG4 + 384 + j];
      float c = sigf(gf) * cst[e] + sigf(gi) * tanhf(gg);
      float h = sigf(go) * tanhf(c);
      cst[e] = c; hbf[e] = (__bf16)h;
      lstm_g[(size_t)(bb * cT + t) * cHID + j] = h;
    }
    __syncthreads();
  }
}

// ---------------- K7: causal attention over T + attended + build FC input ----------------
__global__ __launch_bounds__(256) void
k_caus(const float* __restrict__ x, const float* __restrict__ lstm_g, const float* __restrict__ w_g,
       __bf16* __restrict__ fcin) {
  __shared__ float lst[100 * 128];
  __shared__ float ev[100], ps[100], wl[100];
  __shared__ float mxs;
  int b = blockIdx.x, tid = threadIdx.x;
  for (int e = tid; e < 12800; e += 256) lst[e] = lstm_g[(size_t)b * 12800 + e];
  for (int t = tid; t < 100; t += 256) wl[t] = w_g[b * 100 + t];
  __syncthreads();
  if (tid == 0) { float m = -1e30f; for (int t = 0; t < 100; ++t) m = fmaxf(m, wl[t]); mxs = m; }
  __syncthreads();
  for (int t = tid; t < 100; t += 256) ev[t] = expf(wl[t] - mxs);
  __syncthreads();
  if (tid == 0) { float s = 0.0f; for (int t = 0; t < 100; ++t) { s += ev[t]; ps[t] = s; } }
  __syncthreads();
  for (int e = tid; e < 12800; e += 256) {
    int i = e >> 7, h = e & 127;
    float s = 0.0f;
    for (int j = 0; j <= i; ++j) s += ev[j] * lst[(size_t)j * 128 + h];
    float cur = s / ps[i];
    float att = (i == 0) ? lst[h] : lst[(size_t)i * 128 + h] * 0.5f;
    if (i >= 1 && i <= 98) att += cur * 0.5f;
    fcin[(size_t)(b * cT + i) * cFCK + 250 + h] = (__bf16)att;
  }
  for (int e = tid; e < 100 * 200; e += 256) {
    int i = e / 200, j = e % 200;
    fcin[(size_t)(b * cT + i) * cFCK + j] = (__bf16)x[(size_t)(b * cT + i) * cXROW + j];
  }
  for (int e = tid; e < 100 * 50; e += 256) {
    int i = e / 50, k = e % 50;
    fcin[(size_t)(b * cT + i) * cFCK + 200 + k] = (__bf16)x[(size_t)(b * cT + i) * cXROW + 1280 + k];
  }
  for (int e = tid; e < 100 * 6; e += 256) {
    int i = e / 6, k = e % 6;
    fcin[(size_t)(b * cT + i) * cFCK + 378 + k] = (__bf16)0.0f;
  }
}

// ---------------- K8: final FC (1600x384)x(384x112) + sigmoid ----------------
__global__ __launch_bounds__(256) void
k_fc(const __bf16* __restrict__ fcin, const __bf16* __restrict__ WfcB,
     const float* __restrict__ bfc, float* __restrict__ out) {
  int mt = blockIdx.x, lane = threadIdx.x & 31, wv = threadIdx.x >> 5;
  if (wv >= 7) return;
  const __bf16* arow = fcin + (size_t)(mt * 16 + (lane & 15)) * cFCK;
  const __bf16* brow = WfcB + (size_t)(wv * 16 + (lane & 15)) * cFCK;
  v8f acc = {};
  for (int kt = 0; kt < 12; ++kt) {
    bf16x16 a = load_a_frag(arow, kt * 32, lane);
    bf16x16 b = load_b_frag(brow, kt * 32, lane);
    acc = WMMA_BF16(a, b, acc);
  }
  int n = wv * 16 + (lane & 15);
  if (n < 100) {
    int r0 = mt * 16 + ((lane >= 16) ? 8 : 0);
    float bn = bfc[n];
    #pragma unroll
    for (int vv = 0; vv < 8; ++vv) out[(size_t)(r0 + vv) * 100 + n] = sigf(acc[vv] + bn);
  }
}

// ---------------- host orchestration ----------------
extern "C" void kernel_launch(void* const* d_in, const int* in_sizes, int n_in,
                              void* d_out, int out_size, void* d_ws, size_t ws_size,
                              hipStream_t stream) {
  const float* x        = (const float*)d_in[0];
  const float* node_emb = (const float*)d_in[1];
  const float* path_emb = (const float*)d_in[2];
  const float* Wb       = (const float*)d_in[3];
  const float* bb       = (const float*)d_in[4];
  const float* Wna      = (const float*)d_in[5];
  const float* bna      = (const float*)d_in[6];
  const float* Wia      = (const float*)d_in[7];
  const float* bia      = (const float*)d_in[8];
  const float* Wattn    = (const float*)d_in[9];
  const float* battn    = (const float*)d_in[10];
  const float* kc_emb   = (const float*)d_in[11];
  const float* Wt       = (const float*)d_in[12];
  const float* bt_b     = (const float*)d_in[13];
  const float* Wih      = (const float*)d_in[14];
  const float* Whh      = (const float*)d_in[15];
  const float* bih      = (const float*)d_in[16];
  const float* bhh      = (const float*)d_in[17];
  const float* Wfc      = (const float*)d_in[18];
  const float* bfc      = (const float*)d_in[19];
  float* out = (float*)d_out;

  size_t off = 0;
  auto alloc = [&](size_t bytes) -> void* {
    void* p = (char*)d_ws + off;
    off += (bytes + 255) & ~(size_t)255;
    return p;
  };
  __bf16* WnaB = (__bf16*)alloc((size_t)512 * 512 * 2);
  __bf16* WiaB = (__bf16*)alloc((size_t)112 * 512 * 2);
  __bf16* WbB  = (__bf16*)alloc((size_t)112 * 768 * 2);
  __bf16* WihB = (__bf16*)alloc((size_t)512 * 960 * 2);
  __bf16* WhhB = (__bf16*)alloc((size_t)512 * 128 * 2);
  __bf16* WfcB = (__bf16*)alloc((size_t)112 * 384 * 2);
  float* bna_p = (float*)alloc(512 * 4);
  float* wat_p = (float*)alloc(512 * 4);
  float* bia_p = (float*)alloc(112 * 4);
  float* cw_g  = (float*)alloc((size_t)cBT * 100 * 4);
  float* aw_g  = (float*)alloc((size_t)cBT * 100 * 4);
  __bf16* rnnbf = (__bf16*)alloc((size_t)cBT * cRNNP * 2);
  float* pre_g = (float*)alloc((size_t)cBT * cG4 * 4);
  float* lstm_g= (float*)alloc((size_t)cBT * cHID * 4);
  float* w_g   = (float*)alloc((size_t)cBT * 4);
  __bf16* fcin = (__bf16*)alloc((size_t)cBT * cFCK * 2);

  auto cvt = [&](const float* s, __bf16* d, int r, int c, int rp, int cp) {
    int tot = rp * cp;
    k_cvt<<<(tot + 255) / 256, 256, 0, stream>>>(s, d, r, c, cp, tot);
  };
  cvt(Wna, WnaB, 500, 500, 512, 512);
  cvt(Wia, WiaB, 100, 500, 112, 512);
  cvt(Wb,  WbB,  100, 768, 112, 768);
  cvt(Wih, WihB, 512, 950, 512, 960);
  cvt(Whh, WhhB, 512, 128, 512, 128);
  cvt(Wfc, WfcB, 100, 378, 112, 384);
  k_padf<<<1, 512, 0, stream>>>(bna,   bna_p, 500, 512);
  k_padf<<<1, 512, 0, stream>>>(Wattn, wat_p, 500, 512);
  k_padf<<<1, 128, 0, stream>>>(bia,   bia_p, 100, 112);

  size_t smem1 = (size_t)64 * 512 * 2 + (size_t)(100 * 112 + 112 + 112 + 64 + 112 + 8 * 112 + 8) * 4;
  k1<<<cBT, 256, smem1, stream>>>(x, node_emb, path_emb, WnaB, bna_p, wat_p, battn,
                                  WiaB, bia_p, WbB, bb, kc_emb, Wt, bt_b,
                                  cw_g, rnnbf, w_g);
  k_aw<<<(cBT + 255) / 256, 256, 0, stream>>>(cw_g, aw_g);
  k_na<<<cBT, 256, 0, stream>>>(x, node_emb, path_emb, aw_g, rnnbf);
  k_pre<<<100, 256, 0, stream>>>(rnnbf, WihB, bih, bhh, pre_g);
  k_scan<<<1, 256, 0, stream>>>(pre_g, WhhB, lstm_g);
  k_caus<<<16, 256, 0, stream>>>(x, lstm_g, w_g, fcin);
  k_fc<<<100, 256, 0, stream>>>(fcin, WfcB, bfc, out);
}
